// NMTDecoder_5566277616387
// MI455X (gfx1250) — matrix operand order repaired and checked by reading, not compiled, not run on hardware
//
#include <hip/hip_runtime.h>
#include <hip/hip_bf16.h>

// ---------------------------------------------------------------------------
// NMT decoder step (GRU + attention + vocab classifier) for MI455X (gfx1250).
//
// Strategy (reasoned from MI455X specs, no runtime available):
//  * Dominant cost: classifier GEMM [32,2048]x[2048,32000] per step, 64 steps.
//    fp32 W_cls = 262MB (> 192MB L2, re-streamed from HBM every step: ~0.72ms
//    at 23.3 TB/s). bf16 W_cls = 131MB -> L2-resident across all 64 steps.
//    => one-time fp32->bf16 conversion into workspace, then
//       v_wmma_f32_16x16x32_bf16 with fp32 accumulation.
//  * GRU GEMMs also WMMA-bf16 (weights converted once, 15.7MB, also L2-hot).
//  * Attention/softmax is tiny (8 MFLOP/step) -> plain VALU, one block per b.
//  * Sequential t-loop = 64 x 4 kernel launches on `stream` (capture-safe).
//
// Workspace requirement: ~148.2 MB (layout below, 256B-aligned offsets).
// ---------------------------------------------------------------------------

#define DEVFN __device__ __forceinline__

constexpr int kV  = 32000;           // vocab
constexpr int kE  = 512;             // embedding
constexpr int kH  = 1024;            // hidden
constexpr int kB  = 32;              // batch
constexpr int kS  = 64;              // encoder len
constexpr int kT  = 64;              // target len
constexpr int kTH = 3 * kH;          // 3072 (gate width)
constexpr int kKX = kE + kH;         // 1536 (GRU input width)
constexpr int kU  = 2 * kH;          // 2048 (classifier input width)

typedef __attribute__((ext_vector_type(16))) __bf16 v16bf;
typedef __attribute__((ext_vector_type(8)))  float  v8f;

union FragU { v16bf v; uint4 q[2]; };

// float -> bf16, round-to-nearest-even
DEVFN unsigned short f2bf(float x) {
  unsigned int u = __float_as_uint(x);
  u += 0x7FFFu + ((u >> 16) & 1u);
  return (unsigned short)(u >> 16);
}

// A-matrix fragment (16x32 bf16, ISA 7.12.2): lane-group g holds
// K = {kb+8g .. kb+8g+7} and {kb+16+8g .. kb+23+8g}  (two 16B chunks)
DEVFN v16bf load_frag_a(const unsigned short* row, int kb, int g) {
  FragU f;
  f.q[0] = *reinterpret_cast<const uint4*>(row + kb + g * 8);
  f.q[1] = *reinterpret_cast<const uint4*>(row + kb + 16 + g * 8);
  return f.v;
}

// B-matrix fragment (32x16 bf16): lane-group g holds
// K = {kb+8g .. kb+8g+7} and {kb+16+8g .. kb+23+8g}  (two 16B chunks)
DEVFN v16bf load_frag_b(const unsigned short* row, int kb, int g) {
  FragU f;
  f.q[0] = *reinterpret_cast<const uint4*>(row + kb + g * 8);
  f.q[1] = *reinterpret_cast<const uint4*>(row + kb + 16 + g * 8);
  return f.v;
}

DEVFN v8f wmma_bf16(v16bf a, v16bf b, v8f c) {
  // (neg_a, A, neg_b, B, c_mod, C, reuse_a, reuse_b)
  return __builtin_amdgcn_wmma_f32_16x16x32_bf16(false, a, false, b,
                                                 (short)0, c, false, false);
}

// ---------------------------------------------------------------------------
// One-time prep kernels
// ---------------------------------------------------------------------------
__global__ void cvt_bf16_kernel(const float* __restrict__ src,
                                unsigned short* __restrict__ dst, int n) {
  int stride = gridDim.x * blockDim.x;
  for (int i = blockIdx.x * blockDim.x + threadIdx.x; i < n; i += stride)
    dst[i] = f2bf(src[i]);
}

__global__ void init_state_kernel(const float* __restrict__ h0,
                                  float* __restrict__ h,
                                  float* __restrict__ ctx, int n) {
  int stride = gridDim.x * blockDim.x;
  for (int i = blockIdx.x * blockDim.x + threadIdx.x; i < n; i += stride) {
    h[i]   = h0[i];
    ctx[i] = 0.0f;
  }
}

// ---------------------------------------------------------------------------
// Per-step kernel 1: gather embedding, concat with ctx, bf16-ify inputs
// grid = kB blocks, 256 threads
// ---------------------------------------------------------------------------
__global__ void step_prep_kernel(const float* __restrict__ emb,
                                 const int* __restrict__ tgt,
                                 const float* __restrict__ ctx,
                                 const float* __restrict__ h,
                                 unsigned short* __restrict__ xc,
                                 unsigned short* __restrict__ hb, int t) {
  int b = blockIdx.x, tid = threadIdx.x;
  int y = tgt[b * kT + t];
  const float* er = emb + (size_t)y * kE;
  for (int e = tid; e < kE; e += 256)
    xc[(size_t)b * kKX + e] = f2bf(er[e]);
  for (int i = tid; i < kH; i += 256) {
    xc[(size_t)b * kKX + kE + i] = f2bf(ctx[b * kH + i]);
    hb[(size_t)b * kH + i]       = f2bf(h[b * kH + i]);
  }
}

// ---------------------------------------------------------------------------
// Per-step kernel 2: gate GEMMs via WMMA.
//   gi = xc @ W_ih^T + b_ih   (K=1536)      gh = h @ W_hh^T + b_hh (K=1024)
// 192 N-tiles of 16; each wave does one N-tile x both M-tiles (M=32).
// grid = 24 blocks x 256 threads (8 waves)
// ---------------------------------------------------------------------------
__global__ void gates_kernel(const unsigned short* __restrict__ xc,
                             const unsigned short* __restrict__ hb,
                             const unsigned short* __restrict__ Wih,
                             const unsigned short* __restrict__ Whh,
                             const float* __restrict__ bih,
                             const float* __restrict__ bhh,
                             float* __restrict__ gi, float* __restrict__ gh) {
  int wave = blockIdx.x * 8 + (threadIdx.x >> 5);
  int lane = threadIdx.x & 31;
  int g = lane >> 4, lp = lane & 15;
  int j0 = wave * 16;

  // ---- gi: A = xc [32 x 1536], B rows = W_ih[j0+lp, :] ----
  const unsigned short* a0p = xc + (size_t)lp * kKX;
  const unsigned short* a1p = xc + (size_t)(16 + lp) * kKX;
  const unsigned short* bip = Wih + (size_t)(j0 + lp) * kKX;
  v8f aI0 = {0, 0, 0, 0, 0, 0, 0, 0};
  v8f aI1 = {0, 0, 0, 0, 0, 0, 0, 0};
  for (int kb = 0; kb < kKX; kb += 32) {
    v16bf a0 = load_frag_a(a0p, kb, g);
    v16bf a1 = load_frag_a(a1p, kb, g);
    v16bf bb = load_frag_b(bip, kb, g);
    aI0 = wmma_bf16(a0, bb, aI0);
    aI1 = wmma_bf16(a1, bb, aI1);
  }

  // ---- gh: A = h_bf16 [32 x 1024], B rows = W_hh[j0+lp, :] ----
  const unsigned short* h0p = hb + (size_t)lp * kH;
  const unsigned short* h1p = hb + (size_t)(16 + lp) * kH;
  const unsigned short* bhp = Whh + (size_t)(j0 + lp) * kH;
  v8f aH0 = {0, 0, 0, 0, 0, 0, 0, 0};
  v8f aH1 = {0, 0, 0, 0, 0, 0, 0, 0};
  for (int kb = 0; kb < kH; kb += 32) {
    v16bf a0 = load_frag_a(h0p, kb, g);
    v16bf a1 = load_frag_a(h1p, kb, g);
    v16bf bb = load_frag_b(bhp, kb, g);
    aH0 = wmma_bf16(a0, bb, aH0);
    aH1 = wmma_bf16(a1, bb, aH1);
  }

  float bi  = bih[j0 + lp];
  float bh2 = bhh[j0 + lp];
  for (int j = 0; j < 8; ++j) {
    int m = j + 8 * g;                         // C/D: VGPR j <-> M=j+8g, lane<->N
    gi[(size_t)m * kTH + j0 + lp]        = aI0[j] + bi;
    gi[(size_t)(16 + m) * kTH + j0 + lp] = aI1[j] + bi;
    gh[(size_t)m * kTH + j0 + lp]        = aH0[j] + bh2;
    gh[(size_t)(16 + m) * kTH + j0 + lp] = aH1[j] + bh2;
  }
}

// ---------------------------------------------------------------------------
// Per-step kernel 3: GRU nonlinearity + attention + context.
// grid = kB blocks (one per batch row), 256 threads (8 waves)
// ---------------------------------------------------------------------------
__global__ void gru_attn_kernel(const float* __restrict__ gi,
                                const float* __restrict__ gh,
                                const float* __restrict__ enc,
                                float* __restrict__ h, float* __restrict__ ctx,
                                unsigned short* __restrict__ u,
                                float* __restrict__ attn, int t) {
  __shared__ float h_s[kH];
  __shared__ float sc_s[kS];
  __shared__ float p_s[kS];
  int b = blockIdx.x, tid = threadIdx.x;

  // GRU combine: r,z,n -> h_new
  for (int i = tid; i < kH; i += 256) {
    float gr = gi[(size_t)b * kTH + i]          + gh[(size_t)b * kTH + i];
    float gz = gi[(size_t)b * kTH + kH + i]     + gh[(size_t)b * kTH + kH + i];
    float gin = gi[(size_t)b * kTH + 2 * kH + i];
    float ghn = gh[(size_t)b * kTH + 2 * kH + i];
    float r = 1.0f / (1.0f + expf(-gr));
    float z = 1.0f / (1.0f + expf(-gz));
    float n = tanhf(gin + r * ghn);
    float hn = (1.0f - z) * n + z * h[b * kH + i];
    h_s[i] = hn;
    h[b * kH + i] = hn;
  }
  __syncthreads();

  // scores[s] = <enc[b,s,:], h_new>  (wave-parallel dot + shfl reduction)
  int wave = tid >> 5, lane = tid & 31;
  for (int s = wave; s < kS; s += 8) {
    const float* er = enc + ((size_t)(b * kS + s)) * kH;
    float acc = 0.0f;
    for (int i = lane; i < kH; i += 32) acc += er[i] * h_s[i];
    for (int off = 16; off > 0; off >>= 1) acc += __shfl_xor(acc, off, 32);
    if (lane == 0) sc_s[s] = acc;
  }
  __syncthreads();

  // softmax over S=64 (trivial; serial on one lane) + write attn output
  if (tid == 0) {
    float m = sc_s[0];
    for (int s = 1; s < kS; ++s) m = fmaxf(m, sc_s[s]);
    float sum = 0.0f;
    for (int s = 0; s < kS; ++s) { float e = expf(sc_s[s] - m); p_s[s] = e; sum += e; }
    float inv = 1.0f / sum;
    for (int s = 0; s < kS; ++s) {
      p_s[s] *= inv;
      attn[((size_t)(b * kT + t)) * kS + s] = p_s[s];
    }
  }
  __syncthreads();

  // ctx = p @ enc[b];  u = bf16([ctx, h_new]) for the classifier GEMM
  for (int i = tid; i < kH; i += 256) {
    float acc = 0.0f;
    const float* eb = enc + ((size_t)b * kS) * kH + i;
    for (int s = 0; s < kS; ++s) acc += p_s[s] * eb[(size_t)s * kH];
    ctx[b * kH + i]            = acc;
    u[(size_t)b * kU + i]      = f2bf(acc);
    u[(size_t)b * kU + kH + i] = f2bf(h_s[i]);
  }
}

// ---------------------------------------------------------------------------
// Per-step kernel 4: classifier GEMM  logits[:, t, :] = u @ W_cls^T + b_cls.
// 2000 N-tiles of 16; each wave: one N-tile x both M-tiles, sharing the
// B fragment (W_cls rows) between the two WMMAs.
// grid = 250 blocks x 256 threads (8 waves)
// ---------------------------------------------------------------------------
__global__ void cls_kernel(const unsigned short* __restrict__ u,
                           const unsigned short* __restrict__ Wc,
                           const float* __restrict__ bcls,
                           float* __restrict__ logits, int t) {
  int wave = blockIdx.x * 8 + (threadIdx.x >> 5);
  int lane = threadIdx.x & 31;
  int g = lane >> 4, lp = lane & 15;
  int v0 = wave * 16;

  const unsigned short* arow0 = u + (size_t)lp * kU;
  const unsigned short* arow1 = u + (size_t)(16 + lp) * kU;
  const unsigned short* brow  = Wc + (size_t)(v0 + lp) * kU;

  v8f acc0 = {0, 0, 0, 0, 0, 0, 0, 0};
  v8f acc1 = {0, 0, 0, 0, 0, 0, 0, 0};
  for (int kb = 0; kb < kU; kb += 32) {
    v16bf a0 = load_frag_a(arow0, kb, g);
    v16bf a1 = load_frag_a(arow1, kb, g);
    v16bf bb = load_frag_b(brow, kb, g);
    __builtin_prefetch(brow + kb + 128, 0, 0);   // global_prefetch_b8: stream W_cls
    acc0 = wmma_bf16(a0, bb, acc0);
    acc1 = wmma_bf16(a1, bb, acc1);
  }

  float bias = bcls[v0 + lp];
  size_t col = (size_t)(v0 + lp);
  for (int j = 0; j < 8; ++j) {
    int m0 = j + 8 * g;                                         // batch rows 0..15
    logits[((size_t)(m0 * kT + t)) * kV + col]        = acc0[j] + bias;
    logits[((size_t)((16 + m0) * kT + t)) * kV + col] = acc1[j] + bias;  // rows 16..31
  }
}

// ---------------------------------------------------------------------------
// Host launcher
// ---------------------------------------------------------------------------
extern "C" void kernel_launch(void* const* d_in, const int* in_sizes, int n_in,
                              void* d_out, int out_size, void* d_ws, size_t ws_size,
                              hipStream_t stream) {
  const float* enc   = (const float*)d_in[0];   // [B,S,H]
  const float* h0    = (const float*)d_in[1];   // [B,H]
  const int*   tgt   = (const int*)  d_in[2];   // [B,T]
  const float* emb   = (const float*)d_in[3];   // [V,E]
  const float* W_ih  = (const float*)d_in[4];   // [3H, E+H]
  const float* W_hh  = (const float*)d_in[5];   // [3H, H]
  const float* b_ih  = (const float*)d_in[6];   // [3H]
  const float* b_hh  = (const float*)d_in[7];   // [3H]
  const float* W_cls = (const float*)d_in[8];   // [V, 2H]
  const float* b_cls = (const float*)d_in[9];   // [V]

  float* logits = (float*)d_out;                       // [B,T,V]
  float* attn   = logits + (size_t)kB * kT * kV;       // [B,T,S]

  // Workspace layout (all offsets multiples of 256 bytes); total ~148.2 MB.
  char* ws = (char*)d_ws;
  size_t off = 0;
  unsigned short* Wc_bf  = (unsigned short*)(ws + off); off += (size_t)kV * kU * 2;   // 131,072,000
  unsigned short* Wih_bf = (unsigned short*)(ws + off); off += (size_t)kTH * kKX * 2; //   9,437,184
  unsigned short* Whh_bf = (unsigned short*)(ws + off); off += (size_t)kTH * kH * 2;  //   6,291,456
  float* h_buf   = (float*)(ws + off); off += (size_t)kB * kH * 4;
  float* ctx_buf = (float*)(ws + off); off += (size_t)kB * kH * 4;
  float* gi_buf  = (float*)(ws + off); off += (size_t)kB * kTH * 4;
  float* gh_buf  = (float*)(ws + off); off += (size_t)kB * kTH * 4;
  unsigned short* xc_buf = (unsigned short*)(ws + off); off += (size_t)kB * kKX * 2;
  unsigned short* hb_buf = (unsigned short*)(ws + off); off += (size_t)kB * kH * 2;
  unsigned short* u_buf  = (unsigned short*)(ws + off); off += (size_t)kB * kU * 2;
  (void)off; (void)ws_size; (void)in_sizes; (void)n_in; (void)out_size;

  // One-time (per call) weight conversion fp32 -> bf16: makes W_cls L2-resident.
  cvt_bf16_kernel<<<8192, 256, 0, stream>>>(W_cls, Wc_bf, kV * kU);
  cvt_bf16_kernel<<<2048, 256, 0, stream>>>(W_ih, Wih_bf, kTH * kKX);
  cvt_bf16_kernel<<<1024, 256, 0, stream>>>(W_hh, Whh_bf, kTH * kH);
  init_state_kernel<<<128, 256, 0, stream>>>(h0, h_buf, ctx_buf, kB * kH);

  for (int t = 0; t < kT; ++t) {
    step_prep_kernel<<<kB, 256, 0, stream>>>(emb, tgt, ctx_buf, h_buf,
                                             xc_buf, hb_buf, t);
    gates_kernel<<<kTH / 16 / 8, 256, 0, stream>>>(xc_buf, hb_buf, Wih_bf, Whh_bf,
                                                   b_ih, b_hh, gi_buf, gh_buf);
    gru_attn_kernel<<<kB, 256, 0, stream>>>(gi_buf, gh_buf, enc, h_buf, ctx_buf,
                                            u_buf, attn, t);
    cls_kernel<<<kV / 16 / 8, 256, 0, stream>>>(u_buf, Wc_bf, b_cls, logits, t);
  }
}
